// GlmMoeDsaAttention_22960895164470
// MI455X (gfx1250) — matrix operand order repaired and checked by reading, not compiled, not run on hardware
//
#include <hip/hip_runtime.h>
#include <hip/hip_bf16.h>

// ---------------------------------------------------------------------------
// GLM/DeepSeek MLA + lightning-indexer sparse attention for gfx1250 (MI455X).
// bf16 WMMA everywhere; double-buffered async global->LDS staging (ASYNCcnt).
// ---------------------------------------------------------------------------

#define HID   2048
#define NH    16
#define Q_LORA 768
#define KV_LORA 512
#define QK_NOPE 128
#define QK_ROPE 64
#define QK_HEAD 192
#define V_DIM 128
#define HI_HEADS 8
#define DI_DIM 128
#define TOPK 512
#define SEQ 2048
#define EPSV 1e-5f

typedef __attribute__((ext_vector_type(16))) __bf16 v16bf;
typedef __attribute__((ext_vector_type(8)))  float  v8f;

__device__ __forceinline__ v8f wmma_bf16(v16bf a, v16bf b, v8f c) {
  return __builtin_amdgcn_wmma_f32_16x16x32_bf16(
      false, a, false, b, (short)0, c, false, false);
}

// 16B global -> LDS via CDNA5 async path (ASYNCcnt-tracked).
__device__ __forceinline__ void lds_copy_b128(__bf16* dst, const __bf16* src) {
  unsigned lds_off = (unsigned)(unsigned long long)(const void*)dst;
  asm volatile("global_load_async_to_lds_b128 %0, %1, off"
               :: "v"(lds_off), "v"(src) : "memory");
}
#define ASYNC_WAIT(imm) asm volatile("s_wait_asynccnt " #imm ::: "memory")

// A-fragment (16x32, MxK) from row-major bf16 tile; ISA 7.12.2 layout.
__device__ __forceinline__ v16bf load_frag_a(const __bf16* tile, int ldk, int lane) {
  int m = lane & 15, hi = lane >> 4;
  const __bf16* p = tile + (size_t)m * ldk;
  union { v16bf v; uint4 q[2]; } u;
  u.q[0] = *(const uint4*)(p + 8 * hi);
  u.q[1] = *(const uint4*)(p + 16 + 8 * hi);
  return u.v;
}

// B-fragment (32x16, KxN) from a [N][K] row-major bf16 tile.
__device__ __forceinline__ v16bf load_frag_b(const __bf16* tile, int ldk, int lane) {
  int n = lane & 15, hi = lane >> 4;
  const __bf16* p = tile + (size_t)n * ldk;
  union { v16bf v; uint4 q[2]; } u;
  u.q[0] = *(const uint4*)(p + 8 * hi);
  u.q[1] = *(const uint4*)(p + 16 + 8 * hi);
  return u.v;
}

// ---------------------------------------------------------------------------
// f32 -> bf16 conversion (with scale). n % 8 == 0.
// ---------------------------------------------------------------------------
__global__ __launch_bounds__(256) void k_cvt_bf16(const float* __restrict__ x,
                                                  __bf16* __restrict__ y,
                                                  int n, float alpha) {
  int i = (blockIdx.x * 256 + threadIdx.x) * 8;
  if (i >= n) return;
  float4 a = *(const float4*)(x + i);
  float4 b = *(const float4*)(x + i + 4);
  union { __bf16 h[8]; uint4 q; } u;
  u.h[0] = (__bf16)(a.x * alpha); u.h[1] = (__bf16)(a.y * alpha);
  u.h[2] = (__bf16)(a.z * alpha); u.h[3] = (__bf16)(a.w * alpha);
  u.h[4] = (__bf16)(b.x * alpha); u.h[5] = (__bf16)(b.y * alpha);
  u.h[6] = (__bf16)(b.z * alpha); u.h[7] = (__bf16)(b.w * alpha);
  *(uint4*)(y + i) = u.q;
}

// ---------------------------------------------------------------------------
// GEMM: C[M,N] = alpha * A[M,K] @ B[N,K]^T,  A/B bf16, C f32.
// M%256==0, N%64==0, K%32==0 at all call sites.
// Block 256x64, 8 waves; wave w owns rows [32w,32w+32): 8 WMMAs / k-step.
// Double-buffered async tiles; steady-state loop is branch-free (peeled tail).
// ---------------------------------------------------------------------------
#define GBM 256
#define GBN 64
#define GBK 32

__global__ __launch_bounds__(256) void k_gemm_bf(const __bf16* __restrict__ A,
                                                 const __bf16* __restrict__ B,
                                                 float* __restrict__ C,
                                                 int M, int N, int K, float alpha) {
  __shared__ __align__(16) __bf16 As[2][GBM][GBK];   // 2 x 16 KB
  __shared__ __align__(16) __bf16 Bs[2][GBN][GBK];   // 2 x 4 KB
  int tid = threadIdx.x;
  int wave = tid >> 5, lane = tid & 31;
  int bm = blockIdx.y * GBM, bn = blockIdx.x * GBN;
  int wm = wave * 32;
  v8f acc[2][4] = {};
  int br = tid >> 2, bc = (tid & 3) * 8;

  // A: thread tid fills its whole 32-elem row (4 chunks); B: 1 chunk/thread.
  auto issue_tile = [&](int buf, int k0) {
    const __bf16* ga = A + (size_t)(bm + tid) * K + k0;
    lds_copy_b128(&As[buf][tid][0],  ga);
    lds_copy_b128(&As[buf][tid][8],  ga + 8);
    lds_copy_b128(&As[buf][tid][16], ga + 16);
    lds_copy_b128(&As[buf][tid][24], ga + 24);
    lds_copy_b128(&Bs[buf][br][bc],  B + (size_t)(bn + br) * K + k0 + bc);
  };
  auto compute = [&](int buf) {
    v16bf a0 = load_frag_a(&As[buf][wm][0],      GBK, lane);
    v16bf a1 = load_frag_a(&As[buf][wm + 16][0], GBK, lane);
    v16bf b0 = load_frag_b(&Bs[buf][0][0],  GBK, lane);
    v16bf b1 = load_frag_b(&Bs[buf][16][0], GBK, lane);
    v16bf b2 = load_frag_b(&Bs[buf][32][0], GBK, lane);
    v16bf b3 = load_frag_b(&Bs[buf][48][0], GBK, lane);
    acc[0][0] = wmma_bf16(a0, b0, acc[0][0]);
    acc[0][1] = wmma_bf16(a0, b1, acc[0][1]);
    acc[0][2] = wmma_bf16(a0, b2, acc[0][2]);
    acc[0][3] = wmma_bf16(a0, b3, acc[0][3]);
    acc[1][0] = wmma_bf16(a1, b0, acc[1][0]);
    acc[1][1] = wmma_bf16(a1, b1, acc[1][1]);
    acc[1][2] = wmma_bf16(a1, b2, acc[1][2]);
    acc[1][3] = wmma_bf16(a1, b3, acc[1][3]);
  };

  int nk = K / GBK;
  issue_tile(0, 0);
  for (int t = 0; t + 1 < nk; t++) {        // branch-free steady state
    issue_tile((t + 1) & 1, (t + 1) * GBK);
    ASYNC_WAIT(0x5);                        // drain older 5 copies (tile t)
    __syncthreads();
    compute(t & 1);
    __syncthreads();
  }
  ASYNC_WAIT(0x0);                          // epilogue: last tile
  __syncthreads();
  compute((nk - 1) & 1);

  int n0 = bn + (lane & 15), hi = lane >> 4;
#pragma unroll
  for (int mt = 0; mt < 2; mt++) {
#pragma unroll
    for (int nt = 0; nt < 4; nt++) {
#pragma unroll
      for (int v = 0; v < 8; v++) {
        C[(size_t)(bm + wm + mt * 16 + v + 8 * hi) * N + n0 + nt * 16] =
            alpha * acc[mt][nt][v];
      }
    }
  }
}

// ---------------------------------------------------------------------------
// RMSNorm in-place over rows of length D.
// ---------------------------------------------------------------------------
__global__ __launch_bounds__(256) void k_rmsnorm(float* __restrict__ x,
                                                 const float* __restrict__ w, int D) {
  int s = blockIdx.x, t = threadIdx.x;
  float* xr = x + (size_t)s * D;
  __shared__ float red[8];
  __shared__ float stot;
  float acc = 0.f;
  for (int i = t; i < D; i += 256) { float v = xr[i]; acc += v * v; }
#pragma unroll
  for (int d = 1; d < 32; d <<= 1) acc += __shfl_xor(acc, d);
  if ((t & 31) == 0) red[t >> 5] = acc;
  __syncthreads();
  if (t == 0) { float a = 0.f; for (int i = 0; i < 8; i++) a += red[i]; stot = a; }
  __syncthreads();
  float scale = rsqrtf(stot / (float)D + EPSV);
  for (int i = t; i < D; i += 256) xr[i] = xr[i] * scale * w[i];
}

// ---------------------------------------------------------------------------
// kv post: rmsnorm first 512 -> kvlat, rope last 64 -> krope.
// ---------------------------------------------------------------------------
__global__ __launch_bounds__(256) void k_kv_post(const float* __restrict__ kv,
                                                 const float* __restrict__ w,
                                                 const float* __restrict__ cosb,
                                                 const float* __restrict__ sinb,
                                                 float* __restrict__ kvlat,
                                                 float* __restrict__ krope) {
  int s = blockIdx.x, t = threadIdx.x;
  const float* x = kv + (size_t)s * (KV_LORA + QK_ROPE);
  __shared__ float red[8];
  __shared__ float stot;
  float acc = 0.f;
  for (int i = t; i < KV_LORA; i += 256) { float v = x[i]; acc += v * v; }
#pragma unroll
  for (int d = 1; d < 32; d <<= 1) acc += __shfl_xor(acc, d);
  if ((t & 31) == 0) red[t >> 5] = acc;
  __syncthreads();
  if (t == 0) { float a = 0.f; for (int i = 0; i < 8; i++) a += red[i]; stot = a; }
  __syncthreads();
  float scale = rsqrtf(stot / (float)KV_LORA + EPSV);
  for (int i = t; i < KV_LORA; i += 256)
    kvlat[(size_t)s * KV_LORA + i] = x[i] * scale * w[i];
  if (t < 32) {
    int r = t;
    float x1 = x[KV_LORA + r], x2 = x[KV_LORA + 32 + r];
    float c0 = cosb[(size_t)s * QK_ROPE + r],      sn0 = sinb[(size_t)s * QK_ROPE + r];
    float c1 = cosb[(size_t)s * QK_ROPE + 32 + r], sn1 = sinb[(size_t)s * QK_ROPE + 32 + r];
    krope[(size_t)s * QK_ROPE + r]      = x1 * c0 - x2 * sn0;
    krope[(size_t)s * QK_ROPE + 32 + r] = x2 * c1 + x1 * sn1;
  }
}

// ---------------------------------------------------------------------------
// RoPE on q[:, :, 128:192] in place.
// ---------------------------------------------------------------------------
__global__ __launch_bounds__(512) void k_rope_q(float* __restrict__ q,
                                                const float* __restrict__ cosb,
                                                const float* __restrict__ sinb) {
  int s = blockIdx.x, i = threadIdx.x;
  int h = i >> 5, r = i & 31;
  float* base = q + ((size_t)s * NH + h) * QK_HEAD + QK_NOPE;
  float c0 = cosb[(size_t)s * QK_ROPE + r],      sn0 = sinb[(size_t)s * QK_ROPE + r];
  float c1 = cosb[(size_t)s * QK_ROPE + 32 + r], sn1 = sinb[(size_t)s * QK_ROPE + 32 + r];
  float x1 = base[r], x2 = base[32 + r];
  base[r]      = x1 * c0 - x2 * sn0;
  base[32 + r] = x2 * c1 + x1 * sn1;
}

// ---------------------------------------------------------------------------
// keys_bf[s,h,0:128] = kvb[s,h,0:128]; keys_bf[s,h,128:192] = krope[s].
// ---------------------------------------------------------------------------
__global__ __launch_bounds__(256) void k_build_keys(const float* __restrict__ kvb,
                                                    const float* __restrict__ krope,
                                                    __bf16* __restrict__ keys) {
  int s = blockIdx.x;
  for (int i = threadIdx.x; i < NH * QK_HEAD; i += 256) {
    int h = i / QK_HEAD, d = i % QK_HEAD;
    float v = (d < QK_NOPE)
                  ? kvb[(size_t)s * (NH * 256) + h * 256 + d]
                  : krope[(size_t)s * QK_ROPE + (d - QK_NOPE)];
    keys[(size_t)s * (NH * QK_HEAD) + i] = (__bf16)v;
  }
}

// ---------------------------------------------------------------------------
// Per-head transposed bf16 V: vT[h][c][s] = kvb[s, h, 128+c]. Built once.
// ---------------------------------------------------------------------------
__global__ __launch_bounds__(128) void k_vt(const float* __restrict__ kvb,
                                            __bf16* __restrict__ vT) {
  int s = blockIdx.x, h = blockIdx.y, c = threadIdx.x;
  vT[((size_t)h * V_DIM + c) * SEQ + s] =
      (__bf16)kvb[(size_t)s * (NH * 256) + h * 256 + QK_NOPE + c];
}

// ---------------------------------------------------------------------------
// Indexer key: layer_norm(D=128) then rope dims 0..63. In place safe.
// ---------------------------------------------------------------------------
__global__ __launch_bounds__(128) void k_ik_post(const float* __restrict__ ikraw,
                                                 const float* __restrict__ w,
                                                 const float* __restrict__ b,
                                                 const float* __restrict__ cosb,
                                                 const float* __restrict__ sinb,
                                                 float* __restrict__ ik) {
  int s = blockIdx.x, t = threadIdx.x;
  __shared__ float buf[DI_DIM];
  __shared__ float red[4];
  __shared__ float stats[2];
  float x = ikraw[(size_t)s * DI_DIM + t];
  float m = x;
#pragma unroll
  for (int d = 1; d < 32; d <<= 1) m += __shfl_xor(m, d);
  if ((t & 31) == 0) red[t >> 5] = m;
  __syncthreads();
  if (t == 0) stats[0] = (red[0] + red[1] + red[2] + red[3]) * (1.f / DI_DIM);
  __syncthreads();
  float mean = stats[0];
  float dx = x - mean;
  float v = dx * dx;
#pragma unroll
  for (int d = 1; d < 32; d <<= 1) v += __shfl_xor(v, d);
  __syncthreads();
  if ((t & 31) == 0) red[t >> 5] = v;
  __syncthreads();
  if (t == 0) stats[1] = (red[0] + red[1] + red[2] + red[3]) * (1.f / DI_DIM);
  __syncthreads();
  float var = stats[1];
  float n = dx * rsqrtf(var + EPSV) * w[t] + b[t];
  buf[t] = n;
  __syncthreads();
  float o;
  if (t < QK_ROPE) {
    float c = cosb[(size_t)s * QK_ROPE + t], sn = sinb[(size_t)s * QK_ROPE + t];
    float rot = (t < 32) ? -buf[t + 32] : buf[t - 32];
    o = n * c + rot * sn;
  } else {
    o = n;
  }
  ik[(size_t)s * DI_DIM + t] = o;
}

// ---------------------------------------------------------------------------
// iw[s,h] = hidden[s,:] . idx_weights_w[h,:]
// ---------------------------------------------------------------------------
__global__ __launch_bounds__(256) void k_iw(const float* __restrict__ hidden,
                                            const float* __restrict__ w,
                                            float* __restrict__ iw) {
  int s = blockIdx.x;
  int hv = threadIdx.x >> 5, lane = threadIdx.x & 31;
  const float* x = hidden + (size_t)s * HID;
  const float* ww = w + (size_t)hv * HID;
  float acc = 0.f;
  for (int k = lane * 4; k < HID; k += 128) {
    float4 a = *(const float4*)(x + k);
    float4 b = *(const float4*)(ww + k);
    acc += a.x * b.x + a.y * b.y + a.z * b.z + a.w * b.w;
  }
#pragma unroll
  for (int d = 1; d < 32; d <<= 1) acc += __shfl_xor(acc, d);
  if (lane == 0) iw[(size_t)s * HI_HEADS + hv] = acc;
}

// ---------------------------------------------------------------------------
// ciq[s,d] = sum_h iw_raw[s,h] * rope(iq)[s,h,d]
// ---------------------------------------------------------------------------
__global__ __launch_bounds__(128) void k_ciq(const float* __restrict__ iq,
                                             const float* __restrict__ iw,
                                             const float* __restrict__ cosb,
                                             const float* __restrict__ sinb,
                                             float* __restrict__ ciq) {
  int s = blockIdx.x, d = threadIdx.x;
  bool dorope = d < QK_ROPE;
  float c = 0.f, sn = 0.f;
  if (dorope) { c = cosb[(size_t)s * QK_ROPE + d]; sn = sinb[(size_t)s * QK_ROPE + d]; }
  float acc = 0.f;
#pragma unroll
  for (int hh = 0; hh < HI_HEADS; hh++) {
    const float* base = iq + ((size_t)s * HI_HEADS + hh) * DI_DIM;
    float x = base[d];
    float y;
    if (dorope) {
      float rot = (d < 32) ? -base[d + 32] : base[d - 32];
      y = x * c + rot * sn;
    } else {
      y = x;
    }
    acc += iw[(size_t)s * HI_HEADS + hh] * y;
  }
  ciq[(size_t)s * DI_DIM + d] = acc;
}

// ---------------------------------------------------------------------------
// Per-row top-k -> 2048-bit allow mask.
// ---------------------------------------------------------------------------
__global__ __launch_bounds__(256) void k_topk(const float* __restrict__ iscore,
                                              unsigned int* __restrict__ mask) {
  int q = blockIdx.x, tid = threadIdx.x;
  __shared__ unsigned int su[SEQ];
  __shared__ int cnt;
  for (int k = tid; k < SEQ; k += 256) {
    float f = (k <= q) ? iscore[(size_t)q * SEQ + k] : -__builtin_inff();
    unsigned int u = __float_as_uint(f);
    u = (u & 0x80000000u) ? ~u : (u | 0x80000000u);
    su[k] = u;
  }
  __syncthreads();
  unsigned int thresh = 0;
  if (q + 1 > TOPK) {
    unsigned int t = 0;
    for (int b = 31; b >= 0; --b) {
      unsigned int cand = t | (1u << b);
      if (tid == 0) cnt = 0;
      __syncthreads();
      int c = 0;
      for (int k = tid; k < SEQ; k += 256) c += (su[k] >= cand) ? 1 : 0;
#pragma unroll
      for (int d = 1; d < 32; d <<= 1) c += __shfl_xor(c, d);
      if ((tid & 31) == 0) atomicAdd(&cnt, c);
      __syncthreads();
      if (cnt >= TOPK) t = cand;
      __syncthreads();
    }
    thresh = t;
  }
  int wave = tid >> 5, lane = tid & 31;
  for (int w0 = wave; w0 < SEQ / 32; w0 += 8) {
    int k = w0 * 32 + lane;
    bool pred = (k <= q) && (su[k] >= thresh);
    unsigned long long bal = __ballot(pred);
    if (lane == 0) mask[(size_t)q * (SEQ / 32) + w0] = (unsigned int)bal;
  }
}

// ---------------------------------------------------------------------------
// Masked flash attention: 1 wave per (head, 16-query tile); double-buffered
// async K/V tiles (40 copies/tile/lane -> s_wait_asynccnt 0x28); peeled tail.
// ---------------------------------------------------------------------------
__global__ __launch_bounds__(32) void k_attn(const __bf16* __restrict__ qbf,
                                             const __bf16* __restrict__ kbf,
                                             const __bf16* __restrict__ vT,
                                             const unsigned int* __restrict__ mask,
                                             float* __restrict__ outb) {
  int qt = blockIdx.x;
  int h  = blockIdx.y;
  int lane = threadIdx.x;
  int qbase = qt * 16;

  __shared__ __align__(16) __bf16 Qs[16][QK_HEAD];
  __shared__ __align__(16) __bf16 Ks[2][32][QK_HEAD];
  __shared__ __align__(16) __bf16 Vs[2][V_DIM][32];
  __shared__ __align__(16) __bf16 Ps[16][32];
  __shared__ unsigned int maskw[16];

  for (int i = lane; i < 16 * 24; i += 32) {
    int r = i / 24, c = (i % 24) * 8;
    lds_copy_b128(&Qs[r][c], qbf + ((size_t)(qbase + r) * NH + h) * QK_HEAD + c);
  }
  auto issue_kv = [&](int buf, int kb) {
    int kbase = kb * 32;
    for (int i = lane; i < 32 * 24; i += 32) {
      int r = i / 24, c = (i % 24) * 8;
      lds_copy_b128(&Ks[buf][r][c], kbf + ((size_t)(kbase + r) * NH + h) * QK_HEAD + c);
    }
    for (int i = lane; i < 512; i += 32) {
      int r = i >> 2, j = (i & 3) * 8;
      lds_copy_b128(&Vs[buf][r][j], vT + ((size_t)h * V_DIM + r) * SEQ + kbase + j);
    }
  };
  issue_kv(0, 0);
  ASYNC_WAIT(0x28);   // drains the 12 Q copies (oldest); kv0 may stay in flight
  __syncthreads();

  v16bf qfrag[6];
#pragma unroll
  for (int i = 0; i < 6; i++) qfrag[i] = load_frag_a(&Qs[0][i * 32], QK_HEAD, lane);

  float mrow[8], lrow[8];
  v8f oacc[8];
#pragma unroll
  for (int i = 0; i < 8; i++) { mrow[i] = -1e30f; lrow[i] = 0.f; oacc[i] = (v8f){}; }

  int n = lane & 15, hi = lane >> 4;

  auto compute_block = [&](int buf, int kb) {
    v8f s0 = {}, s1 = {};
#pragma unroll
    for (int i = 0; i < 6; i++) {
      v16bf b0 = load_frag_b(&Ks[buf][0][i * 32],  QK_HEAD, lane);
      v16bf b1 = load_frag_b(&Ks[buf][16][i * 32], QK_HEAD, lane);
      s0 = wmma_bf16(qfrag[i], b0, s0);
      s1 = wmma_bf16(qfrag[i], b1, s1);
    }
    float pmax[8];
#pragma unroll
    for (int v = 0; v < 8; v++) {
      int row = v + 8 * hi;
      int qrow = qbase + row;
      unsigned int mw = maskw[row];
      bool a0 = ((kb * 32 + n)      <= qrow) && ((mw >> n) & 1u);
      bool a1 = ((kb * 32 + 16 + n) <= qrow) && ((mw >> (16 + n)) & 1u);
      float x0 = a0 ? s0[v] : -1e30f;
      float x1 = a1 ? s1[v] : -1e30f;
      s0[v] = x0; s1[v] = x1;
      float mx = fmaxf(x0, x1);
#pragma unroll
      for (int d = 1; d < 16; d <<= 1) mx = fmaxf(mx, __shfl_xor(mx, d));
      pmax[v] = mx;
    }
#pragma unroll
    for (int v = 0; v < 8; v++) {
      float mnew = fmaxf(mrow[v], pmax[v]);
      float corr = __expf(mrow[v] - mnew);
      mrow[v] = mnew;
      float p0 = __expf(s0[v] - mnew);
      float p1 = __expf(s1[v] - mnew);
      float ps = p0 + p1;
#pragma unroll
      for (int d = 1; d < 16; d <<= 1) ps += __shfl_xor(ps, d);
      lrow[v] = lrow[v] * corr + ps;
#pragma unroll
      for (int nt = 0; nt < 8; nt++) oacc[nt][v] *= corr;
      int row = v + 8 * hi;
      Ps[row][n]      = (__bf16)p0;
      Ps[row][16 + n] = (__bf16)p1;
    }
    __syncthreads();
    v16bf pfrag = load_frag_a(&Ps[0][0], 32, lane);
#pragma unroll
    for (int nt = 0; nt < 8; nt++) {
      v16bf vf = load_frag_b(&Vs[buf][nt * 16][0], 32, lane);
      oacc[nt] = wmma_bf16(pfrag, vf, oacc[nt]);
    }
    __syncthreads();
  };

  int kbmax = (qbase + 15) / 32;
  for (int kb = 0; kb < kbmax; kb++) {      // branch-free steady state
    issue_kv((kb + 1) & 1, kb + 1);
    ASYNC_WAIT(0x28);
    if (lane < 16) maskw[lane] = mask[(size_t)(qbase + lane) * 64 + kb];
    __syncthreads();
    compute_block(kb & 1, kb);
  }
  ASYNC_WAIT(0x0);                           // epilogue: last key block
  if (lane < 16) maskw[lane] = mask[(size_t)(qbase + lane) * 64 + kbmax];
  __syncthreads();
  compute_block(kbmax & 1, kbmax);

#pragma unroll
  for (int v = 0; v < 8; v++) {
    int qrow = qbase + v + 8 * hi;
    float inv = 1.0f / lrow[v];
#pragma unroll
    for (int nt = 0; nt < 8; nt++) {
      outb[(size_t)qrow * (NH * V_DIM) + h * V_DIM + nt * 16 + n] = oacc[nt][v] * inv;
    }
  }
}

// ---------------------------------------------------------------------------
// Host launch
// ---------------------------------------------------------------------------
static inline dim3 gemm_grid(int M, int N) {
  return dim3((unsigned)(N / GBN), (unsigned)(M / GBM));
}
static inline unsigned cvt_grid(size_t n) { return (unsigned)((n / 8 + 255) / 256); }

extern "C" void kernel_launch(void* const* d_in, const int* in_sizes, int n_in,
                              void* d_out, int out_size, void* d_ws, size_t ws_size,
                              hipStream_t stream) {
  (void)in_sizes; (void)n_in; (void)out_size; (void)ws_size;
  const float* hidden        = (const float*)d_in[0];
  const float* cosb          = (const float*)d_in[1];
  const float* sinb          = (const float*)d_in[2];
  const float* wq_a          = (const float*)d_in[3];
  const float* q_a_norm_w    = (const float*)d_in[4];
  const float* wq_b          = (const float*)d_in[5];
  const float* wkv_a         = (const float*)d_in[6];
  const float* kv_a_norm_w   = (const float*)d_in[7];
  const float* wkv_b         = (const float*)d_in[8];
  const float* wo            = (const float*)d_in[9];
  const float* idx_wq_b      = (const float*)d_in[10];
  const float* idx_wk        = (const float*)d_in[11];
  const float* idx_k_norm_w  = (const float*)d_in[12];
  const float* idx_k_norm_b  = (const float*)d_in[13];
  const float* idx_weights_w = (const float*)d_in[14];
  float* out = (float*)d_out;

  char* p = (char*)d_ws;
  auto allocf = [&](size_t n) -> float* {
    float* r = (float*)p; p += ((n * 4 + 255) / 256) * 256; return r;
  };
  auto allocb = [&](size_t n) -> __bf16* {
    __bf16* r = (__bf16*)p; p += ((n * 2 + 255) / 256) * 256; return r;
  };
  float* q_lora = allocf((size_t)SEQ * Q_LORA);
  float* qbuf   = allocf((size_t)SEQ * NH * QK_HEAD);
  float* kv     = allocf((size_t)SEQ * (KV_LORA + QK_ROPE));
  float* kvlat  = allocf((size_t)SEQ * KV_LORA);
  float* krope  = allocf((size_t)SEQ * QK_ROPE);
  float* kvb    = allocf((size_t)SEQ * NH * 256);
  float* iq     = allocf((size_t)SEQ * HI_HEADS * DI_DIM);
  float* ikbuf  = allocf((size_t)SEQ * DI_DIM);
  float* iw     = allocf((size_t)SEQ * HI_HEADS);
  float* ciq    = allocf((size_t)SEQ * DI_DIM);
  float* iscore = allocf((size_t)SEQ * SEQ);
  unsigned int* maskb = (unsigned int*)allocf((size_t)SEQ * (SEQ / 32));
  float* attno  = iscore;  // iscore dead after k_topk; alias
  __bf16* hidden_bf   = allocb((size_t)SEQ * HID);
  __bf16* wq_a_bf     = allocb((size_t)Q_LORA * HID);
  __bf16* wq_b_bf     = allocb((size_t)NH * QK_HEAD * Q_LORA);
  __bf16* wkv_a_bf    = allocb((size_t)(KV_LORA + QK_ROPE) * HID);
  __bf16* wkv_b_bf    = allocb((size_t)NH * 256 * KV_LORA);
  __bf16* wo_bf       = allocb((size_t)HID * NH * V_DIM);
  __bf16* idx_wq_b_bf = allocb((size_t)HI_HEADS * DI_DIM * Q_LORA);
  __bf16* idx_wk_bf   = allocb((size_t)DI_DIM * HID);
  __bf16* q_lora_bf   = allocb((size_t)SEQ * Q_LORA);
  __bf16* kvlat_bf    = allocb((size_t)SEQ * KV_LORA);
  __bf16* ik_bf       = allocb((size_t)SEQ * DI_DIM);
  __bf16* ciq_bf      = allocb((size_t)SEQ * DI_DIM);
  __bf16* attno_bf    = allocb((size_t)SEQ * NH * V_DIM);
  __bf16* qbuf_bf     = allocb((size_t)SEQ * NH * QK_HEAD);
  __bf16* keys_bf     = allocb((size_t)SEQ * NH * QK_HEAD);
  __bf16* vT_bf       = allocb((size_t)NH * V_DIM * SEQ);

#define CVT(src, dst, n, a) k_cvt_bf16<<<cvt_grid(n), 256, 0, stream>>>(src, dst, (int)(n), a)

  CVT(hidden, hidden_bf, (size_t)SEQ * HID, 1.0f);
  CVT(wq_a, wq_a_bf, (size_t)Q_LORA * HID, 1.0f);
  CVT(wq_b, wq_b_bf, (size_t)NH * QK_HEAD * Q_LORA, 1.0f);
  CVT(wkv_a, wkv_a_bf, (size_t)(KV_LORA + QK_ROPE) * HID, 1.0f);
  CVT(wkv_b, wkv_b_bf, (size_t)NH * 256 * KV_LORA, 1.0f);
  CVT(wo, wo_bf, (size_t)HID * NH * V_DIM, 1.0f);
  CVT(idx_wq_b, idx_wq_b_bf, (size_t)HI_HEADS * DI_DIM * Q_LORA, 1.0f);
  CVT(idx_wk, idx_wk_bf, (size_t)DI_DIM * HID, 1.0f);

  k_gemm_bf<<<gemm_grid(SEQ, Q_LORA), 256, 0, stream>>>(hidden_bf, wq_a_bf, q_lora, SEQ, Q_LORA, HID, 1.0f);
  k_rmsnorm<<<SEQ, 256, 0, stream>>>(q_lora, q_a_norm_w, Q_LORA);
  CVT(q_lora, q_lora_bf, (size_t)SEQ * Q_LORA, 1.0f);
  k_gemm_bf<<<gemm_grid(SEQ, NH * QK_HEAD), 256, 0, stream>>>(q_lora_bf, wq_b_bf, qbuf, SEQ, NH * QK_HEAD, Q_LORA, 1.0f);
  k_rope_q<<<SEQ, 512, 0, stream>>>(qbuf, cosb, sinb);
  CVT(qbuf, qbuf_bf, (size_t)SEQ * NH * QK_HEAD, 0.07216878364870323f);

  k_gemm_bf<<<gemm_grid(SEQ, KV_LORA + QK_ROPE), 256, 0, stream>>>(hidden_bf, wkv_a_bf, kv, SEQ, KV_LORA + QK_ROPE, HID, 1.0f);
  k_kv_post<<<SEQ, 256, 0, stream>>>(kv, kv_a_norm_w, cosb, sinb, kvlat, krope);
  CVT(kvlat, kvlat_bf, (size_t)SEQ * KV_LORA, 1.0f);
  k_gemm_bf<<<gemm_grid(SEQ, NH * 256), 256, 0, stream>>>(kvlat_bf, wkv_b_bf, kvb, SEQ, NH * 256, KV_LORA, 1.0f);
  k_build_keys<<<SEQ, 256, 0, stream>>>(kvb, krope, keys_bf);
  k_vt<<<dim3(SEQ, NH), 128, 0, stream>>>(kvb, vT_bf);

  k_gemm_bf<<<gemm_grid(SEQ, HI_HEADS * DI_DIM), 256, 0, stream>>>(q_lora_bf, idx_wq_b_bf, iq, SEQ, HI_HEADS * DI_DIM, Q_LORA, 1.0f);
  k_gemm_bf<<<gemm_grid(SEQ, DI_DIM), 256, 0, stream>>>(hidden_bf, idx_wk_bf, ikbuf, SEQ, DI_DIM, HID, 1.0f);
  k_ik_post<<<SEQ, 128, 0, stream>>>(ikbuf, idx_k_norm_w, idx_k_norm_b, cosb, sinb, ikbuf);
  CVT(ikbuf, ik_bf, (size_t)SEQ * DI_DIM, 1.0f);
  k_iw<<<SEQ, 256, 0, stream>>>(hidden, idx_weights_w, iw);
  k_ciq<<<SEQ, 128, 0, stream>>>(iq, iw, cosb, sinb, ciq);
  CVT(ciq, ciq_bf, (size_t)SEQ * DI_DIM, 1.0f);
  k_gemm_bf<<<gemm_grid(SEQ, SEQ), 256, 0, stream>>>(ciq_bf, ik_bf, iscore, SEQ, SEQ, DI_DIM, 0.03125f);
  k_topk<<<SEQ, 256, 0, stream>>>(iscore, maskb);

  k_attn<<<dim3(SEQ / 16, NH), 32, 0, stream>>>(qbuf_bf, keys_bf, vT_bf, maskb, attno);
  CVT(attno, attno_bf, (size_t)SEQ * NH * V_DIM, 1.0f);
  k_gemm_bf<<<gemm_grid(SEQ, HID), 256, 0, stream>>>(attno_bf, wo_bf, out, SEQ, HID, NH * V_DIM, 1.0f);
#undef CVT
}
